// SwinTransformer_65446711657241
// MI455X (gfx1250) — matrix operand order repaired and checked by reading, not compiled, not run on hardware
//
#include <hip/hip_runtime.h>
#include <hip/hip_bf16.h>
#include <math.h>

// ---------------------------------------------------------------------------
// Swin Transformer forward for MI455X (gfx1250, wave32, WMMA).
// GEMMs: f16 operands (pre-converted), f32 accumulate, v_wmma_f32_16x16x32_f16.
// Inner loop: branchless b128 loads, ping-pong double buffering (no reg copies).
// ---------------------------------------------------------------------------

typedef __attribute__((ext_vector_type(16))) _Float16 v16h;
typedef __attribute__((ext_vector_type(8)))  _Float16 v8h;
typedef __attribute__((ext_vector_type(8)))  float    v8f;

__device__ __forceinline__ v16h cat16(v8h lo, v8h hi) {
    return __builtin_shufflevector(lo, hi, 0, 1, 2, 3, 4, 5, 6, 7,
                                   8, 9, 10, 11, 12, 13, 14, 15);
}

__device__ __forceinline__ float gelu_f(float x) {
    return 0.5f * x * (1.0f + erff(x * 0.7071067811865475f));
}

__device__ __forceinline__ float wave_reduce_sum(float v) {
#pragma unroll
    for (int m = 1; m < 32; m <<= 1) v += __shfl_xor(v, m, 32);
    return v;
}
__device__ __forceinline__ float half_reduce_max(float v) {
#pragma unroll
    for (int m = 1; m < 16; m <<= 1) v = fmaxf(v, __shfl_xor(v, m, 32));
    return v;
}
__device__ __forceinline__ float half_reduce_sum(float v) {
#pragma unroll
    for (int m = 1; m < 16; m <<= 1) v += __shfl_xor(v, m, 32);
    return v;
}

__device__ __forceinline__ int imin(int a, int b) { return a < b ? a : b; }

// ---------------------------------------------------------------------------
// f32 -> f16 conversion (weights, once per launch)
// ---------------------------------------------------------------------------
__global__ void swin_f32_to_f16(const float* __restrict__ in,
                                _Float16* __restrict__ out, size_t n) {
    size_t i = blockIdx.x * (size_t)blockDim.x + threadIdx.x;
    if (i < n) out[i] = (_Float16)in[i];
}

// ---------------------------------------------------------------------------
// Patch embedding: 4x4 stride-4 conv, NCHW input -> [B*56*56, 96] rows (NHWC)
// ---------------------------------------------------------------------------
__global__ void swin_patch_embed(const float* __restrict__ x,
                                 const float* __restrict__ w,
                                 const float* __restrict__ b,
                                 float* __restrict__ out, int B) {
    int idx = blockIdx.x * blockDim.x + threadIdx.x;
    int total = B * 56 * 56 * 96;
    if (idx >= total) return;
    int co  = idx % 96;
    int pix = idx / 96;
    int ow  = pix % 56;
    int oh  = (pix / 56) % 56;
    int bb  = pix / (56 * 56);
    float s = b[co];
#pragma unroll
    for (int ci = 0; ci < 3; ++ci)
#pragma unroll
        for (int kh = 0; kh < 4; ++kh)
#pragma unroll
            for (int kw = 0; kw < 4; ++kw)
                s += x[(((size_t)bb * 3 + ci) * 224 + (oh * 4 + kh)) * 224 + (ow * 4 + kw)] *
                     w[((co * 3 + ci) * 4 + kh) * 4 + kw];
    out[idx] = s;
}

// ---------------------------------------------------------------------------
// LayerNorm over last dim. f32 in; f32 out (residual stream) or f16 out
// (GEMM operand). One wave per row.
// ---------------------------------------------------------------------------
__global__ __launch_bounds__(32) void swin_layernorm_f32(
    const float* __restrict__ in, float* __restrict__ out,
    const float* __restrict__ g, const float* __restrict__ be, int C) {
    size_t row = blockIdx.x;
    const float* xr = in + row * (size_t)C;
    float*       yr = out + row * (size_t)C;
    int lane = threadIdx.x;
    float s = 0.0f;
    for (int i = lane; i < C; i += 32) s += xr[i];
    s = wave_reduce_sum(s);
    float mean = s / (float)C;
    float v = 0.0f;
    for (int i = lane; i < C; i += 32) { float d = xr[i] - mean; v += d * d; }
    v = wave_reduce_sum(v) / (float)C;
    float r = rsqrtf(v + 1e-5f);
    for (int i = lane; i < C; i += 32) yr[i] = (xr[i] - mean) * r * g[i] + be[i];
}

__global__ __launch_bounds__(32) void swin_layernorm_f16(
    const float* __restrict__ in, _Float16* __restrict__ out,
    const float* __restrict__ g, const float* __restrict__ be, int C) {
    size_t row = blockIdx.x;
    const float* xr = in + row * (size_t)C;
    _Float16*    yr = out + row * (size_t)C;
    int lane = threadIdx.x;
    float s = 0.0f;
    for (int i = lane; i < C; i += 32) s += xr[i];
    s = wave_reduce_sum(s);
    float mean = s / (float)C;
    float v = 0.0f;
    for (int i = lane; i < C; i += 32) { float d = xr[i] - mean; v += d * d; }
    v = wave_reduce_sum(v) / (float)C;
    float r = rsqrtf(v + 1e-5f);
    for (int i = lane; i < C; i += 32)
        yr[i] = (_Float16)((xr[i] - mean) * r * g[i] + be[i]);
}

// ---------------------------------------------------------------------------
// Window partition (f16 -> f16, optional cyclic shift): [B,H,H,C]->[B*nw*49,C]
// ---------------------------------------------------------------------------
__global__ void swin_windowize_h(const _Float16* __restrict__ in,
                                 _Float16* __restrict__ out,
                                 int B, int H, int C, int w1, int shift) {
    size_t idx = blockIdx.x * (size_t)blockDim.x + threadIdx.x;
    size_t total = (size_t)B * H * H * C;
    if (idx >= total) return;
    int c = (int)(idx % C);
    size_t r = idx / C;
    int p = (int)(r % 49);
    size_t r2 = r / 49;
    int nw = w1 * w1;
    int w_ = (int)(r2 % nw);
    int b  = (int)(r2 / nw);
    int wi = w_ / w1, wj = w_ % w1, pi = p / 7, pj = p % 7;
    int h  = (wi * 7 + pi + shift) % H;
    int ww = (wj * 7 + pj + shift) % H;
    out[idx] = in[(((size_t)b * H + h) * H + ww) * C + c];
}

// Inverse: windowed f32 [B*nw*49, C] scattered (+= residual) into X [B,H,H,C]
__global__ void swin_scatter_add(const float* __restrict__ win, float* __restrict__ X,
                                 int B, int H, int C, int w1, int shift) {
    size_t idx = blockIdx.x * (size_t)blockDim.x + threadIdx.x;
    size_t total = (size_t)B * H * H * C;
    if (idx >= total) return;
    int c = (int)(idx % C);
    size_t r = idx / C;
    int p = (int)(r % 49);
    size_t r2 = r / 49;
    int nw = w1 * w1;
    int w_ = (int)(r2 % nw);
    int b  = (int)(r2 / nw);
    int wi = w_ / w1, wj = w_ % w1, pi = p / 7, pj = p % 7;
    int h  = (wi * 7 + pi + shift) % H;
    int ww = (wj * 7 + pj + shift) % H;
    X[(((size_t)b * H + h) * H + ww) * C + c] += win[idx];
}

// ---------------------------------------------------------------------------
// Patch merging permute: [B,H,H,C] -> [B*(H/2)*(H/2), 4C], col=(wp*2+hp)*C+c
// ---------------------------------------------------------------------------
__global__ void swin_merge_permute(const float* __restrict__ in, float* __restrict__ out,
                                   int B, int H, int C) {
    size_t idx = blockIdx.x * (size_t)blockDim.x + threadIdx.x;
    size_t total = (size_t)B * H * H * C;
    if (idx >= total) return;
    int H2 = H / 2;
    int C4 = 4 * C;
    int oc = (int)(idx % C4);
    size_t r = idx / C4;
    int w2 = (int)(r % H2);
    int h2 = (int)((r / H2) % H2);
    int b  = (int)(r / ((size_t)H2 * H2));
    int c  = oc % C;
    int pp = oc / C;
    int wp = pp >> 1;
    int hp = pp & 1;
    out[idx] = in[(((size_t)b * H + (2 * h2 + hp)) * H + (2 * w2 + wp)) * C + c];
}

// ---------------------------------------------------------------------------
// Mean pool over 49 spatial positions: f16 [B*49, C] -> f16 [B, C]
// ---------------------------------------------------------------------------
__global__ void swin_meanpool_h(const _Float16* __restrict__ in,
                                _Float16* __restrict__ out, int B, int C) {
    int idx = blockIdx.x * blockDim.x + threadIdx.x;
    if (idx >= B * C) return;
    int c = idx % C;
    int b = idx / C;
    float s = 0.0f;
    for (int p = 0; p < 49; ++p) s += (float)in[((size_t)b * 49 + p) * C + c];
    out[idx] = (_Float16)(s * (1.0f / 49.0f));
}

// ---------------------------------------------------------------------------
// WMMA GEMM: D[M,N] = act(A[M,K] @ W[N,K]^T + bias) (+ residual)
//   A, W are f16; accumulate f32; output f32 (outF) or f16 (outH).
// One wave per 16x64 strip (4 accumulators, A fragment reused 4x).
// REQUIRES K % 32 == 0. Out-of-range rows/cols clamped on load (pollutes only
// masked rows/cols of D). Ping-pong double buffering: two fragment sets
// alternate as load-target / compute-source, so no rotation copies.
//
// CDNA5 layouts (ISA 7.12.2):
//   A(16x32 f16): row=lane%16, elem t -> K=half*16+t (chosen here, same as B)
//   B(32x16 f16): col=lane%16, K half selected by lane/16, elem t -> K=half*16+t (chosen here, same as A)
//   C/D(16x16 f32): elem i -> row i+8*(lane/16), col lane%16
// ---------------------------------------------------------------------------
struct GemmFrag {
    v8h a0, a1;
    v8h b0[4], b1[4];
};

__device__ __forceinline__ void gemm_load_frag(GemmFrag& f, const v8h* arow,
                                               const v8h* const* brow, int step) {
    int j = step * 4;  // step*32 halfs -> v8h index step*4
    f.a0 = arow[j];
    f.a1 = arow[j + 1];
#pragma unroll
    for (int nt = 0; nt < 4; ++nt) {
        f.b0[nt] = brow[nt][j];
        f.b1[nt] = brow[nt][j + 1];
    }
}

__device__ __forceinline__ void gemm_mma(v8f acc[4], const GemmFrag& f) {
    v16h a = cat16(f.a0, f.a1);
#pragma unroll
    for (int nt = 0; nt < 4; ++nt)
        acc[nt] = __builtin_amdgcn_wmma_f32_16x16x32_f16(
            false, a, false, cat16(f.b0[nt], f.b1[nt]), (short)0, acc[nt], false, false);
}

__global__ __launch_bounds__(32) void swin_gemm_wmma(
    const _Float16* __restrict__ A, const _Float16* __restrict__ W,
    const float* __restrict__ bias, const float* __restrict__ res,
    float* __restrict__ outF, _Float16* __restrict__ outH,
    int M, int N, int K, int act) {
    int tn = blockIdx.x, tm = blockIdx.y;
    int lane = threadIdx.x;
    int half = lane >> 4;
    int l16  = lane & 15;

    int row  = tm * 16 + l16;
    int rowc = imin(row, M - 1);
    // v8h index j covers halfs [8j .. 8j+7] from the per-lane base
    const v8h* arow = reinterpret_cast<const v8h*>(A + (size_t)rowc * K + half * 16);
    const v8h* brow[4];
#pragma unroll
    for (int nt = 0; nt < 4; ++nt) {
        int colc = imin(tn * 64 + nt * 16 + l16, N - 1);
        brow[nt] = reinterpret_cast<const v8h*>(W + (size_t)colc * K + half * 16);
    }

    v8f acc[4] = {{}, {}, {}, {}};
    int nsteps = K >> 5;

    GemmFrag f0, f1;
    gemm_load_frag(f0, arow, brow, 0);
    for (int step = 1; step < nsteps; step += 2) {
        gemm_load_frag(f1, arow, brow, step);   // loads in flight over mma(f0)
        gemm_mma(acc, f0);
        if (step + 1 < nsteps)
            gemm_load_frag(f0, arow, brow, step + 1);  // in flight over mma(f1)
        gemm_mma(acc, f1);
    }
    if (nsteps & 1)
        gemm_mma(acc, f0);   // trailing odd step (loaded by last loop iteration
                             // or by the prologue when nsteps == 1)

#pragma unroll
    for (int nt = 0; nt < 4; ++nt) {
        int c = tn * 64 + nt * 16 + l16;
#pragma unroll
        for (int i = 0; i < 8; ++i) {
            int r = tm * 16 + i + half * 8;
            if (r < M && c < N) {
                float v = acc[nt][i];
                if (bias) v += bias[c];
                if (act == 1) v = gelu_f(v);
                if (outH) {
                    outH[(size_t)r * N + c] = (_Float16)v;
                } else {
                    if (res) v += res[(size_t)r * N + c];
                    outF[(size_t)r * N + c] = v;
                }
            }
        }
    }
}

// ---------------------------------------------------------------------------
// Windowed MSA core. One wave per (batch, window, head). f16 QKV in, f16 out.
// QKV: [B*nw*49, 3C]; q cols [h*32,..), k at C+h*32, v at 2C+h*32.
// 49 padded to 64 via index clamping:
//  - S columns q>=49 forced to -inf pre-softmax (clamped garbage harmless)
//  - V rows q>=49 multiplied by exactly-zero probabilities
//  - Q rows p>=49 pollute only discarded output rows
// 1/sqrt(32) applied post-WMMA in the f32 softmax epilogue.
// ---------------------------------------------------------------------------
__global__ __launch_bounds__(32) void swin_attention(
    const _Float16* __restrict__ QKV, const float* __restrict__ rel,
    _Float16* __restrict__ O, int nw, int w1, int nh, int C, int shifted) {
    __shared__ __align__(16) _Float16 lds[16 * 64];

    int bi   = blockIdx.x;
    int head = bi % nh;
    int w_   = (bi / nh) % nw;
    int b    = bi / (nh * nw);
    int lane = threadIdx.x;
    int half = lane >> 4;
    int l16  = lane & 15;

    size_t baseRow = ((size_t)b * nw + w_) * 49;
    int ld   = 3 * C;
    int qoff = head * 32;
    int koff = C + head * 32;
    int voff = 2 * C + head * 32;
    const float scale = 0.17677669529663687f; // 32^-0.5

    // B tiles of K^T: 16 contiguous halfs per lane (2x b128)
    v16h bK[4];
#pragma unroll
    for (int nt = 0; nt < 4; ++nt) {
        int qc = imin(nt * 16 + l16, 48);
        const v8h* kp = reinterpret_cast<const v8h*>(QKV + (baseRow + qc) * ld + koff + half * 16);
        bK[nt] = cat16(kp[0], kp[1]);
    }
    // B tiles of V (strided gather across rows; clamped, unconditional)
    v16h bV[2][2];
#pragma unroll
    for (int kt = 0; kt < 2; ++kt)
#pragma unroll
        for (int nt2 = 0; nt2 < 2; ++nt2) {
            int c = nt2 * 16 + l16;
#pragma unroll
            for (int t = 0; t < 16; ++t) {
                int qc = imin(kt * 32 + half * 16 + t, 48);
                bV[kt][nt2][t] = QKV[(baseRow + qc) * ld + voff + c];
            }
        }

    int wr = w_ / w1, wc = w_ % w1;

    for (int mt = 0; mt < 4; ++mt) {
        // A tile of Q (rows mt*16..+15, K = head_dim 32)
        int pc = imin(mt * 16 + l16, 48);
        const v8h* qp = reinterpret_cast<const v8h*>(QKV + (baseRow + pc) * ld + qoff + half * 16);
        v16h aQ = cat16(qp[0], qp[1]);

        v8f s[4];
#pragma unroll
        for (int nt = 0; nt < 4; ++nt) {
            v8f z = {};
            s[nt] = __builtin_amdgcn_wmma_f32_16x16x32_f16(false, aQ, false, bK[nt],
                                                           (short)0, z, false, false);
        }
        // scale, rel bias, shift mask, row softmax (row = 16 lanes x 4 tiles)
#pragma unroll
        for (int i = 0; i < 8; ++i) {
            int r = mt * 16 + i + half * 8;
            int ri  = (r < 49) ? r : 0;
            int pi2 = ri / 7, pj2 = ri % 7;
            float mx = -INFINITY;
#pragma unroll
            for (int nt = 0; nt < 4; ++nt) {
                int q_ = nt * 16 + l16;
                float v = s[nt][i] * scale;
                if (q_ < 49 && r < 49) {
                    int qi2 = q_ / 7, qj2 = q_ % 7;
                    v += rel[(head * 13 + (pi2 - qi2 + 6)) * 13 + (pj2 - qj2 + 6)];
                    if (shifted) {
                        bool m = false;
                        if (wr == w1 - 1) m = m || ((pi2 < 4) != (qi2 < 4));
                        if (wc == w1 - 1) m = m || ((pj2 < 4) != (qj2 < 4));
                        if (m) v = -INFINITY;
                    }
                } else {
                    v = -INFINITY;
                }
                s[nt][i] = v;
                mx = fmaxf(mx, v);
            }
            mx = half_reduce_max(mx);
            float sum = 0.0f;
#pragma unroll
            for (int nt = 0; nt < 4; ++nt) {
                float sv = s[nt][i];
                float e_ = (sv == -INFINITY) ? 0.0f : expf(sv - mx);
                s[nt][i] = e_;
                sum += e_;
            }
            sum = half_reduce_sum(sum);
            float inv = (sum > 0.0f) ? 1.0f / sum : 0.0f;
#pragma unroll
            for (int nt = 0; nt < 4; ++nt) s[nt][i] *= inv;
        }
        // stage probabilities to LDS [16 rows x 64 cols] f16
#pragma unroll
        for (int nt = 0; nt < 4; ++nt)
#pragma unroll
            for (int i = 0; i < 8; ++i) {
                int rloc = i + half * 8;
                int cc   = nt * 16 + l16;
                lds[rloc * 64 + cc] = (_Float16)s[nt][i];
            }
        __syncthreads();
        // reload in WMMA A layout: row = lane%16, K(e,half)
        v16h aP[2];
#pragma unroll
        for (int kt = 0; kt < 2; ++kt)
#pragma unroll
            for (int e = 0; e < 16; ++e) {
                int cc = kt * 32 + half * 16 + e;
                aP[kt][e] = lds[l16 * 64 + cc];
            }
        // out = P @ V (K = 64 padded window positions)
#pragma unroll
        for (int nt2 = 0; nt2 < 2; ++nt2) {
            v8f o = {};
            o = __builtin_amdgcn_wmma_f32_16x16x32_f16(false, aP[0], false, bV[0][nt2],
                                                       (short)0, o, false, false);
            o = __builtin_amdgcn_wmma_f32_16x16x32_f16(false, aP[1], false, bV[1][nt2],
                                                       (short)0, o, false, false);
#pragma unroll
            for (int i = 0; i < 8; ++i) {
                int r = mt * 16 + i + half * 8;
                if (r < 49)
                    O[(baseRow + r) * C + head * 32 + nt2 * 16 + l16] = (_Float16)o[i];
            }
        }
        __syncthreads();
    }
}

// ---------------------------------------------------------------------------
// Host orchestration
// ---------------------------------------------------------------------------
struct SwinBlk {
    const float *ln1_g, *ln1_b, *qkv_w, *qkv_b, *rel, *proj_w, *proj_b;
    const float *ln2_g, *ln2_b, *fc1_w, *fc1_b, *fc2_w, *fc2_b;
    const _Float16 *qkv_w16, *proj_w16, *fc1_w16, *fc2_w16;
};
struct SwinStage {
    const float *mg, *mb, *mw;
    const _Float16 *mw16;
    SwinBlk blk[6];
};

extern "C" void kernel_launch(void* const* d_in, const int* in_sizes, int n_in,
                              void* d_out, int out_size, void* d_ws, size_t ws_size,
                              hipStream_t stream) {
    (void)in_sizes; (void)n_in; (void)out_size; (void)ws_size;
    // Params flattened in setup_inputs() dict insertion order.
    int idx = 0;
    auto in = [&](void) { return (const float*)d_in[idx++]; };

    const float* x       = in();
    const float* patch_w = in();
    const float* patch_b = in();
    const float* ln0_g   = in();
    const float* ln0_b   = in();

    const int nblk[4] = {2, 2, 6, 2};
    SwinStage stages[4];
    for (int s = 0; s < 4; ++s) {
        if (s > 0) {
            stages[s].mg = in();
            stages[s].mb = in();
            stages[s].mw = in();
        } else {
            stages[s].mg = stages[s].mb = stages[s].mw = nullptr;
        }
        for (int bI = 0; bI < nblk[s]; ++bI) {
            SwinBlk& bp = stages[s].blk[bI];
            bp.ln1_g  = in(); bp.ln1_b  = in();
            bp.qkv_w  = in(); bp.qkv_b  = in();
            bp.rel    = in();
            bp.proj_w = in(); bp.proj_b = in();
            bp.ln2_g  = in(); bp.ln2_b  = in();
            bp.fc1_w  = in(); bp.fc1_b  = in();
            bp.fc2_w  = in(); bp.fc2_b  = in();
        }
    }
    const float* nl_g    = in();
    const float* nl_b    = in();
    const float* mlp1_w  = in();
    const float* mlp1_b  = in();
    const float* mlp2_w  = in();
    const float* mlp2_b  = in();
    const float* cls_w   = in();
    const float* cls_b   = in();

    // Workspace carve-out
    const int B = 32;
    const size_t XCAP  = (size_t)B * 56 * 56 * 96;   // max activation elems
    const size_t T2CAP = XCAP * 4;                   // qkv / fc1 elems
    char* ws = (char*)d_ws;
    float*    X   = (float*)ws;    ws += XCAP * sizeof(float);
    float*    T1  = (float*)ws;    ws += XCAP * sizeof(float);
    _Float16* T1h = (_Float16*)ws; ws += XCAP * sizeof(_Float16);
    _Float16* T3h = (_Float16*)ws; ws += XCAP * sizeof(_Float16);
    _Float16* T4h = (_Float16*)ws; ws += XCAP * sizeof(_Float16);
    _Float16* T2h = (_Float16*)ws; ws += T2CAP * sizeof(_Float16);
    _Float16* P0h = (_Float16*)ws; ws += (size_t)B * 768 * sizeof(_Float16);
    _Float16* P1h = (_Float16*)ws; ws += (size_t)B * 1536 * sizeof(_Float16);
    _Float16* P2h = (_Float16*)ws; ws += (size_t)B * 768 * sizeof(_Float16);
    _Float16* WB  = (_Float16*)ws; // f16 weight pool (~61 MB), cursor below

    // Pre-convert all GEMM weights to f16 (loop-invariant within a launch)
    _Float16* wcur = WB;
    auto conv = [&](const float* src, size_t n) -> const _Float16* {
        _Float16* dst = wcur;
        wcur += n;
        swin_f32_to_f16<<<(int)((n + 255) / 256), 256, 0, stream>>>(src, dst, n);
        return dst;
    };
    {
        int d = 96;
        for (int s = 0; s < 4; ++s) {
            if (s > 0) {
                stages[s].mw16 = conv(stages[s].mw, (size_t)(2 * d) * (4 * d));
                d *= 2;
            } else {
                stages[s].mw16 = nullptr;
            }
            for (int bI = 0; bI < nblk[s]; ++bI) {
                SwinBlk& bp = stages[s].blk[bI];
                bp.qkv_w16  = conv(bp.qkv_w, (size_t)(3 * d) * d);
                bp.proj_w16 = conv(bp.proj_w, (size_t)d * d);
                bp.fc1_w16  = conv(bp.fc1_w, (size_t)(4 * d) * d);
                bp.fc2_w16  = conv(bp.fc2_w, (size_t)d * (4 * d));
            }
        }
    }
    const _Float16* mlp1_w16 = conv(mlp1_w, (size_t)1536 * 768);
    const _Float16* mlp2_w16 = conv(mlp2_w, (size_t)768 * 1536);
    const _Float16* cls_w16  = conv(cls_w, (size_t)1000 * 768);

    auto gemm = [&](const _Float16* A, const _Float16* W, const float* bias,
                    const float* res, float* outF, _Float16* outH,
                    int M, int N, int K, int act) {
        dim3 grid((N + 63) / 64, (M + 15) / 16);
        swin_gemm_wmma<<<grid, 32, 0, stream>>>(A, W, bias, res, outF, outH,
                                                M, N, K, act);
    };

    // Patch embed + ln0 (f32 residual stream)
    {
        int total = B * 56 * 56 * 96;
        swin_patch_embed<<<(total + 255) / 256, 256, 0, stream>>>(x, patch_w, patch_b, X, B);
        swin_layernorm_f32<<<B * 56 * 56, 32, 0, stream>>>(X, X, ln0_g, ln0_b, 96);
    }

    int H = 56, C = 96;
    for (int s = 0; s < 4; ++s) {
        if (s > 0) {
            size_t total = (size_t)B * H * H * C;
            swin_merge_permute<<<(int)((total + 255) / 256), 256, 0, stream>>>(X, T1, B, H, C);
            int H2 = H / 2;
            swin_layernorm_f16<<<B * H2 * H2, 32, 0, stream>>>(T1, T3h, stages[s].mg, stages[s].mb, 4 * C);
            gemm(T3h, stages[s].mw16, nullptr, nullptr, X, nullptr, B * H2 * H2, 2 * C, 4 * C, 0);
            H = H2;
            C = 2 * C;
        }
        int w1 = H / 7, nw = w1 * w1, nh = C / 32;
        int rows = B * H * H;                 // == B*nw*49
        size_t total = (size_t)rows * C;
        for (int bI = 0; bI < nblk[s]; ++bI) {
            SwinBlk& bp = stages[s].blk[bI];
            int shifted = (bI % 2 == 1 && H > 7) ? 1 : 0;
            int shift   = shifted ? 3 : 0;
            // attention branch
            swin_layernorm_f16<<<rows, 32, 0, stream>>>(X, T1h, bp.ln1_g, bp.ln1_b, C);
            swin_windowize_h<<<(int)((total + 255) / 256), 256, 0, stream>>>(T1h, T3h, B, H, C, w1, shift);
            gemm(T3h, bp.qkv_w16, bp.qkv_b, nullptr, nullptr, T2h, rows, 3 * C, C, 0);
            swin_attention<<<B * nw * nh, 32, 0, stream>>>(T2h, bp.rel, T4h, nw, w1, nh, C, shifted);
            gemm(T4h, bp.proj_w16, bp.proj_b, nullptr, T1, nullptr, rows, C, C, 0);
            swin_scatter_add<<<(int)((total + 255) / 256), 256, 0, stream>>>(T1, X, B, H, C, w1, shift);
            // MLP branch
            swin_layernorm_f16<<<rows, 32, 0, stream>>>(X, T1h, bp.ln2_g, bp.ln2_b, C);
            gemm(T1h, bp.fc1_w16, bp.fc1_b, nullptr, nullptr, T2h, rows, 4 * C, C, 1);
            gemm(T2h, bp.fc2_w16, bp.fc2_b, X, X, nullptr, rows, C, 4 * C, 0);
        }
    }

    // Head: LN -> mean pool -> MLP(gelu) -> classifier
    swin_layernorm_f16<<<B * 49, 32, 0, stream>>>(X, T1h, nl_g, nl_b, 768);
    swin_meanpool_h<<<(B * 768 + 255) / 256, 256, 0, stream>>>(T1h, P0h, B, 768);
    gemm(P0h, mlp1_w16, mlp1_b, nullptr, nullptr, P1h, B, 1536, 768, 1);
    gemm(P1h, mlp2_w16, mlp2_b, nullptr, nullptr, P2h, B, 768, 1536, 0);
    gemm(P2h, cls_w16, cls_b, nullptr, (float*)d_out, nullptr, B, 1000, 768, 0);
}
